// HGTDetector_39822936769061
// MI455X (gfx1250) — compile-verified, matrix operand order checked by reading, not run on hardware
//
#include <hip/hip_runtime.h>
#include <hip/hip_bf16.h>
#include <math.h>
#include <stdint.h>

typedef __attribute__((ext_vector_type(16))) _Float16 v16h;
typedef __attribute__((ext_vector_type(8)))  _Float16 v8h;
typedef __attribute__((ext_vector_type(8)))  float    v8f;

#define D 128
#define LRELU_SLOPE 0.01f

// ---------------------------------------------------------------------------
// Ordered-float <-> uint mapping for atomic segment max
// ---------------------------------------------------------------------------
__device__ __forceinline__ unsigned f2ord(float f) {
  unsigned u = __float_as_uint(f);
  return (u & 0x80000000u) ? ~u : (u | 0x80000000u);
}
__device__ __forceinline__ float ord2f(unsigned u) {
  return (u & 0x80000000u) ? __uint_as_float(u & 0x7fffffffu) : __uint_as_float(~u);
}

// ---------------------------------------------------------------------------
// f16 fragment loader (CDNA5 wave32 layout, CDNA5 ISA §7.12.2).
// Matrix X is row-major f16 with row stride K (multiple of 32, rows 16B
// aligned). Per lane: row r, halves k=kbase..kbase+7 and kbase+16..kbase+23
// -> exactly two global_load_b128.
// ---------------------------------------------------------------------------
__device__ __forceinline__ v16h load_frag_f16(const _Float16* __restrict__ X,
                                              int strideK, int r, int kbase) {
  const _Float16* p = X + (size_t)r * (size_t)strideK + kbase;
  v8h lo = *(const v8h*)p;
  v8h hi = *(const v8h*)(p + 16);
  v16h f;
#pragma unroll
  for (int h = 0; h < 8; ++h) { f[h] = lo[h]; f[h + 8] = hi[h]; }
  return f;
}

__device__ __forceinline__ v8f wmma_step(v16h a, v16h b, v8f c) {
  return __builtin_amdgcn_wmma_f32_16x16x32_f16(
      /*neg_a=*/false, a, /*neg_b=*/false, b,
      /*c_mod=*/(short)0, c, /*reuse_a=*/false, /*reuse_b=*/false);
}

// ---------------------------------------------------------------------------
// Tiled GEMM: C = act(A[M,K] @ B[K,N] + bias), B supplied TRANSPOSED (N x K).
// One wave per 16x64 C strip: one A fragment feeds 4 independent WMMA
// accumulator chains (4x A-reuse, ~12.8 FLOP/B, ILP across acc chains).
// K fully unrolled through v_wmma_f32_16x16x32_f16.
// C32 (f32) and C16 (f16) outputs each optional.
// ACT: 0 = identity, 1 = leaky_relu(0.01). Requires N % 64 == 0.
// ---------------------------------------------------------------------------
template <int ACT, int K>
__global__ void gemm_wmma_kernel(const _Float16* __restrict__ A,
                                 const _Float16* __restrict__ Bt,
                                 const float* __restrict__ bias,
                                 float* __restrict__ C32,
                                 _Float16* __restrict__ C16,
                                 int ldc, int M, int N) {
  int gtid = blockIdx.x * blockDim.x + threadIdx.x;
  int wave = gtid >> 5;
  int lane = threadIdx.x & 31;
  int ngroups = N >> 6;                 // 64-wide N groups
  int mt = wave / ngroups;
  int ng = wave - mt * ngroups;
  int m0 = mt * 16;
  int n0 = ng * 64;
  if (m0 >= M) return;  // whole wave exits together; EXEC all-1 inside

  int ra = m0 + (lane & 15);            // A row for this lane
  int rb = n0 + (lane & 15);            // first Bt row (B column) for this lane
  int kb = (lane >> 4) << 3;            // +8 K-offset for lanes 16..31

  v8f acc0 = {}, acc1 = {}, acc2 = {}, acc3 = {};
#pragma unroll
  for (int k0 = 0; k0 < K; k0 += 32) {
    if ((k0 & 255) == 0 && k0 + 256 < K) {
      __builtin_prefetch(A + (size_t)ra * K + k0 + 256, 0, 1);  // global_prefetch_b8
    }
    v16h af = load_frag_f16(A, K, ra, k0 + kb);
    v16h b0 = load_frag_f16(Bt, K, rb +  0, k0 + kb);
    v16h b1 = load_frag_f16(Bt, K, rb + 16, k0 + kb);
    v16h b2 = load_frag_f16(Bt, K, rb + 32, k0 + kb);
    v16h b3 = load_frag_f16(Bt, K, rb + 48, k0 + kb);
    acc0 = wmma_step(af, b0, acc0);
    acc1 = wmma_step(af, b1, acc1);
    acc2 = wmma_step(af, b2, acc2);
    acc3 = wmma_step(af, b3, acc3);
  }

  int mbase = (lane >> 4) << 3;         // C rows: VGPR r -> M = r + 8*(lane>=16)
#pragma unroll
  for (int t = 0; t < 4; ++t) {
    v8f acc = (t == 0) ? acc0 : (t == 1) ? acc1 : (t == 2) ? acc2 : acc3;
    int n = n0 + t * 16 + (lane & 15);
    float bv = bias ? bias[n] : 0.0f;
#pragma unroll
    for (int r = 0; r < 8; ++r) {
      size_t m = (size_t)(m0 + r + mbase);
      float v = acc[r] + bv;
      if (ACT == 1) v = (v > 0.0f) ? v : LRELU_SLOPE * v;
      if (C32) C32[m * (size_t)ldc + n] = v;
      if (C16) C16[m * (size_t)ldc + n] = (_Float16)v;
    }
  }
}

// ---------------------------------------------------------------------------
// Staging: f32 -> f16 packed copies
// ---------------------------------------------------------------------------
// user_x[:, 9:777] -> packed NU x 768 f16
__global__ void conv_des_kernel(const float* __restrict__ user_x,
                                _Float16* __restrict__ des16, int NU) {
  size_t i = (size_t)blockIdx.x * blockDim.x + threadIdx.x;
  size_t n = (size_t)NU * 768;
  if (i >= n) return;
  size_t u = i / 768;
  size_t j = i - u * 768;
  des16[i] = (_Float16)user_x[u * 777 + 9 + j];
}

// W (K x N f32, row-major) -> Wt (N x K f16, row-major)
__global__ void w_transpose_kernel(const float* __restrict__ W,
                                   _Float16* __restrict__ Wt, int K, int N) {
  int i = blockIdx.x * blockDim.x + threadIdx.x;
  if (i >= K * N) return;
  int k = i / N;
  int n = i - k * N;
  Wt[(size_t)n * K + k] = (_Float16)W[i];
}

// ---------------------------------------------------------------------------
// Tiny encoders: cat (4->32) and num (5->32) with leaky relu, f16 out
// ---------------------------------------------------------------------------
__global__ void enc_small_kernel(const float* __restrict__ user_x,
                                 const float* __restrict__ Wc, const float* __restrict__ bc,
                                 const float* __restrict__ Wn, const float* __restrict__ bn,
                                 _Float16* __restrict__ enc16, int NU) {
  int i = blockIdx.x * blockDim.x + threadIdx.x;
  if (i >= NU * 64) return;
  int u = i >> 6;
  int c = i & 63;
  const float* row = user_x + (size_t)u * 777;
  float acc;
  if (c < 32) {
    acc = bc[c];
#pragma unroll
    for (int k = 0; k < 4; ++k) acc += row[k] * Wc[k * 32 + c];
  } else {
    int cc = c - 32;
    acc = bn[cc];
#pragma unroll
    for (int k = 0; k < 5; ++k) acc += row[4 + k] * Wn[k * 32 + cc];
  }
  acc = (acc > 0.0f) ? acc : LRELU_SLOPE * acc;
  enc16[(size_t)u * D + c] = (_Float16)acc;
}

// ---------------------------------------------------------------------------
// Edge attention phase (wave per edge, coalesced 128-float rows)
// ---------------------------------------------------------------------------
__global__ void edge_logits_kernel(const int* __restrict__ src, const int* __restrict__ dst,
                                   const float* __restrict__ KA, const float* __restrict__ Q,
                                   const float* __restrict__ p_scalar, float scale,
                                   float* __restrict__ logit, unsigned* __restrict__ segmax,
                                   int E) {
  int gtid = blockIdx.x * blockDim.x + threadIdx.x;
  int e = gtid >> 5;
  int lane = threadIdx.x & 31;
  if (e >= E) return;
  int s = src[e], d = dst[e];
  const float* ka = KA + (size_t)s * D;
  const float* q  = Q  + (size_t)d * D;
  float sum = 0.0f;
#pragma unroll
  for (int j = 0; j < 4; ++j) sum += ka[lane + 32 * j] * q[lane + 32 * j];
#pragma unroll
  for (int off = 16; off > 0; off >>= 1) sum += __shfl_xor(sum, off, 32);
  if (lane == 0) {
    float l = sum * p_scalar[0] * scale;
    logit[e] = l;
    atomicMax(&segmax[d], f2ord(l));
  }
}

__global__ void edge_exp_kernel(const int* __restrict__ dst,
                                const unsigned* __restrict__ segmax,
                                float* __restrict__ logit, float* __restrict__ segsum,
                                int E) {
  int e = blockIdx.x * blockDim.x + threadIdx.x;
  if (e >= E) return;
  int d = dst[e];
  float m = ord2f(segmax[d]);
  float ex = __expf(logit[e] - m);
  logit[e] = ex;
  atomicAdd(&segsum[d], ex);
}

__global__ void edge_scatter_kernel(const int* __restrict__ src, const int* __restrict__ dst,
                                    const float* __restrict__ VM,
                                    const float* __restrict__ logit_exp,
                                    const float* __restrict__ segsum,
                                    float* __restrict__ agg, int E) {
  int gtid = blockIdx.x * blockDim.x + threadIdx.x;
  int e = gtid >> 5;
  int lane = threadIdx.x & 31;
  if (e >= E) return;
  int s = src[e], d = dst[e];
  float alpha = logit_exp[e] / segsum[d];
  const float* vm = VM + (size_t)s * D;
  float* out = agg + (size_t)d * D;
#pragma unroll
  for (int j = 0; j < 4; ++j)
    atomicAdd(&out[lane + 32 * j], vm[lane + 32 * j] * alpha);
}

// ---------------------------------------------------------------------------
// Elementwise: exact gelu (f32 -> f16), skip mix (-> f16), 2-class softmax head
// ---------------------------------------------------------------------------
__global__ void gelu_kernel(const float* __restrict__ x, _Float16* __restrict__ y, size_t n) {
  size_t i = (size_t)blockIdx.x * blockDim.x + threadIdx.x;
  if (i >= n) return;
  float v = x[i];
  y[i] = (_Float16)(0.5f * v * (1.0f + erff(v * 0.70710678118654752440f)));
}

__global__ void mix_kernel(const float* __restrict__ ou, const float* __restrict__ hu,
                           const float* __restrict__ skip_scalar,
                           _Float16* __restrict__ h2u16, size_t n) {
  size_t i = (size_t)blockIdx.x * blockDim.x + threadIdx.x;
  if (i >= n) return;
  float su = 1.0f / (1.0f + __expf(-skip_scalar[0]));
  h2u16[i] = (_Float16)(su * ou[i] + (1.0f - su) * hu[i]);
}

__global__ void head_kernel(const float* __restrict__ hid,
                            const float* __restrict__ W2, const float* __restrict__ b2,
                            float* __restrict__ out, int NU) {
  int gtid = blockIdx.x * blockDim.x + threadIdx.x;
  int u = gtid >> 5;
  int lane = threadIdx.x & 31;
  if (u >= NU) return;
  const float* h = hid + (size_t)u * D;
  float a0 = 0.0f, a1 = 0.0f;
#pragma unroll
  for (int j = 0; j < 4; ++j) {
    int k = lane + 32 * j;
    float hv = h[k];
    a0 += hv * W2[k * 2 + 0];
    a1 += hv * W2[k * 2 + 1];
  }
#pragma unroll
  for (int off = 16; off > 0; off >>= 1) {
    a0 += __shfl_xor(a0, off, 32);
    a1 += __shfl_xor(a1, off, 32);
  }
  if (lane == 0) {
    float z0 = a0 + b2[0], z1 = a1 + b2[1];
    float m = fmaxf(z0, z1);
    float e0 = __expf(z0 - m), e1 = __expf(z1 - m);
    float inv = 1.0f / (e0 + e1);
    out[(size_t)u * 2 + 0] = e0 * inv;
    out[(size_t)u * 2 + 1] = e1 * inv;
  }
}

// ---------------------------------------------------------------------------
// Launcher
// ---------------------------------------------------------------------------
template <int ACT, int K>
static void launch_gemm(const _Float16* A, const _Float16* Bt, const float* bias,
                        float* C32, _Float16* C16, int ldc, int M, int N,
                        hipStream_t stream) {
  long long waves = (long long)((M + 15) / 16) * (N / 64);  // 16x64 strip per wave
  int blocks = (int)((waves * 32 + 255) / 256);
  gemm_wmma_kernel<ACT, K><<<blocks, 256, 0, stream>>>(A, Bt, bias, C32, C16, ldc, M, N);
}

static inline void* align_up(void* p, size_t a) {
  return (void*)(((uintptr_t)p + a - 1) & ~(uintptr_t)(a - 1));
}

extern "C" void kernel_launch(void* const* d_in, const int* in_sizes, int n_in,
                              void* d_out, int out_size, void* d_ws, size_t ws_size,
                              hipStream_t stream) {
  const float* user_x = (const float*)d_in[0];
  const int*   e_f    = (const int*)d_in[2];
  const int*   e_r    = (const int*)d_in[3];
  const int NU = in_sizes[0] / 777;
  const int EF = in_sizes[2] / 2;
  const int ER = in_sizes[3] / 2;
  const int* sf = e_f;  const int* df = e_f + EF;
  const int* sr = e_r;  const int* dr = e_r + ER;

  // params, setup_inputs() insertion order, (w,b) tuples flattened w-then-b
  const float* enc_u_cat_w = (const float*)d_in[5];
  const float* enc_u_cat_b = (const float*)d_in[6];
  const float* enc_u_num_w = (const float*)d_in[7];
  const float* enc_u_num_b = (const float*)d_in[8];
  const float* enc_u_des_w = (const float*)d_in[9];
  const float* enc_u_des_b = (const float*)d_in[10];
  const float* enc_u_out_w = (const float*)d_in[11];
  const float* enc_u_out_b = (const float*)d_in[12];
  const float* k_u_w  = (const float*)d_in[15];
  const float* k_u_b  = (const float*)d_in[16];
  const float* q_u_w  = (const float*)d_in[17];
  const float* q_u_b  = (const float*)d_in[18];
  const float* v_u_w  = (const float*)d_in[19];
  const float* v_u_b  = (const float*)d_in[20];
  const float* out_u_w = (const float*)d_in[21];
  const float* out_u_b = (const float*)d_in[22];
  const float* skip_u  = (const float*)d_in[23];
  const float* A_fm = (const float*)d_in[33];
  const float* M_fm = (const float*)d_in[34];
  const float* p_f  = (const float*)d_in[35];
  const float* A_rm = (const float*)d_in[36];
  const float* M_rm = (const float*)d_in[37];
  const float* p_r  = (const float*)d_in[38];
  const float* cls1_w = (const float*)d_in[42];
  const float* cls1_b = (const float*)d_in[43];
  const float* cls2_w = (const float*)d_in[44];
  const float* cls2_b = (const float*)d_in[45];

  // ---------------- workspace layout ----------------
  size_t NU128 = (size_t)NU * D;
  float* F = (float*)d_ws;             // 7 x (NU x 128) f32
  float* F0 = F + 0 * NU128;  // hu
  float* F1 = F + 1 * NU128;  // q_u   -> hid
  float* F2 = F + 2 * NU128;  // kA_f  -> ou
  float* F3 = F + 3 * NU128;  // kA_r
  float* F4 = F + 4 * NU128;  // vM_f
  float* F5 = F + 5 * NU128;  // vM_r
  float* F6 = F + 6 * NU128;  // agg
  _Float16* H = (_Float16*)align_up(F + 7 * NU128, 64);  // NU x 768 f16 region
  _Float16* des16 = H;                        // dead after des GEMM
  _Float16* hu16  = H + 0 * NU128;            // reuse of des16 space
  _Float16* k16   = H + 1 * NU128;
  _Float16* v16   = H + 2 * NU128;
  _Float16* agg16 = H + 3 * NU128;
  _Float16* h2u16 = H + 4 * NU128;
  _Float16* enc16 = (_Float16*)align_up(H + (size_t)NU * 768, 64);  // NU x 128 f16
  float* lf = (float*)align_up(enc16 + NU128, 64);
  float* lr = lf + EF;
  unsigned* segmax = (unsigned*)(lr + ER);
  float*    segsum = (float*)(segmax + NU);
  _Float16* wts = (_Float16*)align_up(segsum + NU, 64);  // transposed f16 weights
  _Float16* desWt  = wts;                    // 64 x 768
  _Float16* eoWt   = desWt + 64 * 768;       // 128 x 128 each below
  _Float16* kWt    = eoWt  + D * D;
  _Float16* qWt    = kWt   + D * D;
  _Float16* vWt    = qWt   + D * D;
  _Float16* ouWt   = vWt   + D * D;
  _Float16* AfWt   = ouWt  + D * D;
  _Float16* ArWt   = AfWt  + D * D;
  _Float16* MfWt   = ArWt  + D * D;
  _Float16* MrWt   = MfWt  + D * D;
  _Float16* c1Wt   = MrWt  + D * D;

  const float scale = 0.08838834764831845f;  // 1/sqrt(128)
  float* outp = (float*)d_out;

  // 0. stage weights (f16, transposed) and the description block (f16 packed)
  auto wt = [&](const float* W, _Float16* Wt, int K, int N) {
    w_transpose_kernel<<<(K * N + 255) / 256, 256, 0, stream>>>(W, Wt, K, N);
  };
  wt(enc_u_des_w, desWt, 768, 64);
  wt(enc_u_out_w, eoWt, D, D);
  wt(k_u_w, kWt, D, D);  wt(q_u_w, qWt, D, D);  wt(v_u_w, vWt, D, D);
  wt(out_u_w, ouWt, D, D);
  wt(A_fm, AfWt, D, D);  wt(A_rm, ArWt, D, D);
  wt(M_fm, MfWt, D, D);  wt(M_rm, MrWt, D, D);
  wt(cls1_w, c1Wt, D, D);
  {
    size_t n = (size_t)NU * 768;
    conv_des_kernel<<<(int)((n + 255) / 256), 256, 0, stream>>>(user_x, des16, NU);
  }

  // 1. user encoder
  enc_small_kernel<<<(NU * 64 + 255) / 256, 256, 0, stream>>>(
      user_x, enc_u_cat_w, enc_u_cat_b, enc_u_num_w, enc_u_num_b, enc16, NU);
  launch_gemm<1, 768>(des16, desWt, enc_u_des_b, nullptr, enc16 + 64, D, NU, 64, stream);
  launch_gemm<1, 128>(enc16, eoWt, enc_u_out_b, F0, hu16, D, NU, D, stream);  // hu

  // 2. k/q/v projections (k,v only needed as f16 GEMM inputs; q as f32)
  launch_gemm<0, 128>(hu16, kWt, k_u_b, nullptr, k16, D, NU, D, stream);
  launch_gemm<0, 128>(hu16, qWt, q_u_b, F1, nullptr, D, NU, D, stream);
  launch_gemm<0, 128>(hu16, vWt, v_u_b, nullptr, v16, D, NU, D, stream);

  // 3. relation transforms (f32 outputs for edge gathers)
  launch_gemm<0, 128>(k16, AfWt, nullptr, F2, nullptr, D, NU, D, stream);  // kA_f
  launch_gemm<0, 128>(k16, ArWt, nullptr, F3, nullptr, D, NU, D, stream);  // kA_r
  launch_gemm<0, 128>(v16, MfWt, nullptr, F4, nullptr, D, NU, D, stream);  // vM_f
  launch_gemm<0, 128>(v16, MrWt, nullptr, F5, nullptr, D, NU, D, stream);  // vM_r

  // 4. segment softmax accumulators (re-init every call; capture-safe)
  hipMemsetAsync(segmax, 0, (size_t)NU * sizeof(unsigned), stream);
  hipMemsetAsync(segsum, 0, (size_t)NU * sizeof(float), stream);
  hipMemsetAsync(F6, 0, NU128 * sizeof(float), stream);

  // 5. edge attention: combined segmax -> exp/segsum -> weighted scatter
  edge_logits_kernel<<<(EF * 32 + 255) / 256, 256, 0, stream>>>(sf, df, F2, F1, p_f, scale, lf, segmax, EF);
  edge_logits_kernel<<<(ER * 32 + 255) / 256, 256, 0, stream>>>(sr, dr, F3, F1, p_r, scale, lr, segmax, ER);
  edge_exp_kernel<<<(EF + 255) / 256, 256, 0, stream>>>(df, segmax, lf, segsum, EF);
  edge_exp_kernel<<<(ER + 255) / 256, 256, 0, stream>>>(dr, segmax, lr, segsum, ER);
  edge_scatter_kernel<<<(EF * 32 + 255) / 256, 256, 0, stream>>>(sf, df, F4, lf, segsum, F6, EF);
  edge_scatter_kernel<<<(ER * 32 + 255) / 256, 256, 0, stream>>>(sr, dr, F5, lr, segsum, F6, ER);

  // 6. output projection + skip mix
  gelu_kernel<<<(int)((NU128 + 255) / 256), 256, 0, stream>>>(F6, agg16, NU128);
  launch_gemm<0, 128>(agg16, ouWt, out_u_b, F2, nullptr, D, NU, D, stream);  // ou
  mix_kernel<<<(int)((NU128 + 255) / 256), 256, 0, stream>>>(F2, F0, skip_u, h2u16, NU128);

  // 7. classifier head
  launch_gemm<1, 128>(h2u16, c1Wt, cls1_b, F1, nullptr, D, NU, D, stream);  // hid
  head_kernel<<<(NU * 32 + 255) / 256, 256, 0, stream>>>(F1, cls2_w, cls2_b, outp, NU);
}
